// BertMoeLayer_65395172049425
// MI455X (gfx1250) — matrix operand order repaired and checked by reading, not compiled, not run on hardware
//
#include <hip/hip_runtime.h>
#include <hip/hip_bf16.h>
#include <stdint.h>

#define B_ 8
#define S_ 1024
#define H_ 768
#define NH_ 12
#define DH_ 64
#define F_ 3072
#define E_ 8
#define T_ (B_*S_)

typedef __attribute__((ext_vector_type(16))) __bf16 v16bf;
typedef __attribute__((ext_vector_type(8)))  float  v8f;

union Frag { uint32_t u[8]; v16bf v; };

__device__ __forceinline__ unsigned short f2bf(float f) {
  union { float f; uint32_t u; } a; a.f = f;
  uint32_t u = a.u;
  uint32_t r = (u + 0x7FFFu + ((u >> 16) & 1u)) >> 16;  // RNE
  return (unsigned short)r;
}
__device__ __forceinline__ uint32_t pack2f(float lo, float hi) {
  return ((uint32_t)f2bf(hi) << 16) | (uint32_t)f2bf(lo);
}
__device__ __forceinline__ unsigned short toBF(float f) { return f2bf(f); }
__device__ __forceinline__ unsigned short toBF(unsigned short h) { return h; }

// A-matrix 16x32 bf16 K-pattern per ISA 7.12.2 (hi = lane>>4)
__device__ __forceinline__ int kpatA(int i, int hi) {
  return (i < 4) ? (hi * 8 + 2 * i) : (16 + hi * 8 + 2 * (i - 4));
}
// B-matrix 32x16 bf16 K-pattern (same K order as A)
__device__ __forceinline__ int kpatB(int i, int hi) { return kpatA(i, hi); }

__device__ __forceinline__ float gelu_exact(float x) {
  return 0.5f * x * (1.0f + erff(x * 0.70710678118654752f));
}
__device__ __forceinline__ v8f wmma_bf16(const Frag& a, const Frag& b, v8f c) {
  return __builtin_amdgcn_wmma_f32_16x16x32_bf16(false, a.v, false, b.v,
                                                 (short)0, c, false, false);
}

// CDNA5 async global->LDS copy (16 bytes/lane), tracked by ASYNCcnt.
__device__ __forceinline__ void async_copy16(uint32_t lds_byte_addr, const void* gptr) {
  asm volatile("global_load_async_to_lds_b128 %0, %1, off"
               :: "v"(lds_byte_addr), "v"((uint64_t)(uintptr_t)gptr)
               : "memory");
}
__device__ __forceinline__ void wait_async0() {
  asm volatile("s_wait_asynccnt 0" ::: "memory");
}

// ---------------------------------------------------------------------------
// One-time weight prep: fp32 [R][C] -> bf16 [C][R] (per blockIdx.z matrix).
// ---------------------------------------------------------------------------
__global__ __launch_bounds__(256) void transpose_bf16_kernel(
    const float* __restrict__ in, unsigned short* __restrict__ out, int R, int C)
{
  __shared__ float tile[32][33];
  const size_t zo = (size_t)blockIdx.z * (size_t)R * (size_t)C;
  const int c0 = blockIdx.x * 32, r0 = blockIdx.y * 32;
  const int tx = threadIdx.x & 31, ty = threadIdx.x >> 5;
  #pragma unroll
  for (int i = 0; i < 4; i++)
    tile[ty + i * 8][tx] = in[zo + (size_t)(r0 + ty + i * 8) * C + c0 + tx];
  __syncthreads();
  #pragma unroll
  for (int i = 0; i < 4; i++)
    out[zo + (size_t)(c0 + ty + i * 8) * R + r0 + tx] = f2bf(tile[tx][ty + i * 8]);
}

// ---------------------------------------------------------------------------
// Generic WMMA GEMM: C[M,N] = epilogue(A[M,K] @ B[K,N] + bias [+ resid])
// B supplied PRE-TRANSPOSED bf16: Bt[N][K]. B tiles staged to LDS with
// global_load_async_to_lds_b128. Block 64x128, 8 waves, 32x32/wave, K-step 32.
// OUTM: 0 = fp32, 1 = bf16, 2 = bf16 scattered to [B][NH][DH][S] (V layout).
// ---------------------------------------------------------------------------
#define BM 64
#define BN 128
#define BKK 32
#define APAD 34   // 68B row stride (4B aligned), conflict-free
#define BPAD 40   // 80B row stride (16B aligned for async b128), conflict-free

template<typename TA, int ACT, bool RES, int OUTM,
         bool GATHER_A, bool SCATTER_C, bool EXPERT>
__global__ __launch_bounds__(256) void wmma_gemm_kernel(
    const TA* __restrict__ A, const unsigned short* __restrict__ Bt,
    const float* __restrict__ bias, const float* __restrict__ Resid,
    void* __restrict__ Cout, const int* __restrict__ idx,
    const int* __restrict__ seg_off, const int* __restrict__ seg_cnt,
    int M, int N, int K, long strideB, int strideBias)
{
  __shared__ unsigned short As[BM * APAD];
  __shared__ alignas(16) unsigned short Bs[BN * BPAD];

  const int tid = threadIdx.x;
  int z = 0, segOff = 0, segCnt = M;
  if (EXPERT) { z = blockIdx.z; segOff = seg_off[z]; segCnt = seg_cnt[z]; }
  const int rowTile = blockIdx.y * BM;
  if (rowTile >= segCnt) return;
  int valid = segCnt - rowTile; if (valid > BM) valid = BM;
  const int n0 = blockIdx.x * BN;

  const unsigned short* Bz = Bt + (size_t)z * (size_t)strideB;
  const float* biasz = bias + (size_t)z * (size_t)strideBias;

  const int lane = tid & 31, wid = tid >> 5;
  const int wm = wid & 1, wn = wid >> 1;
  const int hi = lane >> 4, l15 = lane & 15;

  v8f acc[2][2];
  #pragma unroll
  for (int a0 = 0; a0 < 2; a0++)
    #pragma unroll
    for (int a1 = 0; a1 < 2; a1++) acc[a0][a1] = v8f{};

  for (int k0 = 0; k0 < K; k0 += BKK) {
    __syncthreads();
    // B tile: 128 rows (N) x 32 ushorts (K) = 512 x 16B chunks, async DMA
    #pragma unroll
    for (int c = 0; c < 2; c++) {
      int ch = tid + c * 256;
      int n = ch >> 2, q = ch & 3;
      async_copy16((uint32_t)(uintptr_t)&Bs[n * BPAD + q * 8],
                   Bz + (size_t)(n0 + n) * K + k0 + q * 8);
    }
    if (k0 + BKK < K)
      __builtin_prefetch(Bz + (size_t)(n0 + (tid & 127)) * K + k0 + BKK, 0, 1);
    // A tile (64x32), convert to bf16 (synchronous; small vs B)
    #pragma unroll
    for (int i = 0; i < 8; i++) {
      int e = tid + i * 256;          // 0..2047
      int m = e >> 5, kk = e & 31;
      unsigned short v = 0;
      if (m < valid) {
        int g  = segOff + rowTile + m;
        int ar = GATHER_A ? idx[g] : g;
        v = toBF(A[(size_t)ar * K + k0 + kk]);
      }
      As[m * APAD + kk] = v;
    }
    wait_async0();
    __syncthreads();

    Frag af[2], bfg[2];
    #pragma unroll
    for (int sm = 0; sm < 2; sm++) {
      int r = wm * 32 + sm * 16 + l15;
      #pragma unroll
      for (int i = 0; i < 8; i++)
        af[sm].u[i] = *(const uint32_t*)&As[r * APAD + kpatA(i, hi)];
    }
    #pragma unroll
    for (int sn = 0; sn < 2; sn++) {
      int c = wn * 32 + sn * 16 + l15;
      #pragma unroll
      for (int i = 0; i < 8; i++)
        bfg[sn].u[i] = *(const uint32_t*)&Bs[c * BPAD + kpatB(i, hi)];
    }
    #pragma unroll
    for (int sm = 0; sm < 2; sm++)
      #pragma unroll
      for (int sn = 0; sn < 2; sn++)
        acc[sm][sn] = wmma_bf16(af[sm], bfg[sn], acc[sm][sn]);
  }

  // epilogue
  #pragma unroll
  for (int sm = 0; sm < 2; sm++) {
    #pragma unroll
    for (int sn = 0; sn < 2; sn++) {
      int gc = n0 + wn * 32 + sn * 16 + l15;
      #pragma unroll
      for (int j = 0; j < 8; j++) {
        int lr = wm * 32 + sm * 16 + j + 8 * hi;
        if (lr >= valid) continue;
        int g    = segOff + rowTile + lr;
        int crow = SCATTER_C ? idx[g] : g;
        float val = acc[sm][sn][j] + biasz[gc];
        if (RES) val += Resid[(size_t)crow * N + gc];
        if (ACT == 1) val = gelu_exact(val);
        if (OUTM == 2) {           // V layout: [B][NH][DH][S] bf16
          int bb = crow >> 10, ss = crow & 1023;
          int hh = gc >> 6,  dd = gc & 63;
          ((unsigned short*)Cout)[(((size_t)bb * NH_ + hh) * DH_ + dd) * S_ + ss]
              = f2bf(val);
        } else if (OUTM == 1) {
          ((unsigned short*)Cout)[(size_t)crow * N + gc] = f2bf(val);
        } else {
          ((float*)Cout)[(size_t)crow * N + gc] = val;
        }
      }
    }
  }
}

// ---------------------------------------------------------------------------
// Attention: 1 wave per (b, h, 16-query tile). Scores for all 1024 keys in
// LDS, softmax in-wave (probs mirrored to padded bf16 LDS), probs @ V WMMA.
// V is in [B][NH][DH][S] bf16 so P.V B-fragments are contiguous along keys.
// ---------------------------------------------------------------------------
#define SPS 1032  // bf16 prob row stride (pad: 2064B rows -> conflict-free)

__global__ __launch_bounds__(32) void attn_kernel(
    const unsigned short* __restrict__ Qb, const unsigned short* __restrict__ Kb,
    const unsigned short* __restrict__ Vt, float* __restrict__ ctx)
{
  __shared__ float sc[16 * 1024];            // 64 KB scores
  __shared__ unsigned short sp[16 * SPS];    // 33 KB bf16 probs
  const int lane = threadIdx.x;
  const int qt = blockIdx.x, h = blockIdx.y, b = blockIdx.z;
  const int hi = lane >> 4, l15 = lane & 15;

  // token t's row base in [B,S,NH,DH] bf16
  #define TOKBASE(t) (((size_t)(b * S_ + (t))) * H_ + h * DH_)

  // Q fragments for the two K-steps over DH=64
  Frag aq[2];
  {
    int q = qt * 16 + l15;
    size_t qb = TOKBASE(q);
    #pragma unroll
    for (int s = 0; s < 2; s++)
      #pragma unroll
      for (int i = 0; i < 8; i++)
        aq[s].u[i] = *(const uint32_t*)(Qb + qb + s * 32 + kpatA(i, hi));
  }

  // phase 1: scores = Q @ K^T * 1/sqrt(64)
  for (int kt = 0; kt < 64; kt++) {
    Frag bk[2];
    int key = kt * 16 + l15;            // B col = lane%16
    size_t kb = TOKBASE(key);
    #pragma unroll
    for (int s = 0; s < 2; s++)
      #pragma unroll
      for (int i = 0; i < 8; i++)
        bk[s].u[i] = *(const uint32_t*)(Kb + kb + s * 32 + kpatB(i, hi));
    v8f c = v8f{};
    c = wmma_bf16(aq[0], bk[0], c);
    c = wmma_bf16(aq[1], bk[1], c);
    #pragma unroll
    for (int j = 0; j < 8; j++)
      sc[(j + 8 * hi) * 1024 + kt * 16 + l15] = c[j] * 0.125f;
  }
  __syncthreads();

  // phase 2: softmax; lanes l and l^16 own the two halves of row l%16
  {
    int r = l15, c0 = hi * 512;
    float m = -3.4e38f;
    for (int c = 0; c < 512; c++) m = fmaxf(m, sc[r * 1024 + c0 + c]);
    m = fmaxf(m, __shfl_xor(m, 16, 32));
    float ssum = 0.f;
    for (int c = 0; c < 512; c++) {
      float e = __expf(sc[r * 1024 + c0 + c] - m);
      sc[r * 1024 + c0 + c] = e;
      ssum += e;
    }
    ssum += __shfl_xor(ssum, 16, 32);
    float inv = 1.0f / ssum;
    for (int c = 0; c < 512; c += 2) {
      float p0 = sc[r * 1024 + c0 + c] * inv;
      float p1 = sc[r * 1024 + c0 + c + 1] * inv;
      *(uint32_t*)&sp[r * SPS + c0 + c] = pack2f(p0, p1);
    }
  }
  __syncthreads();

  // phase 3: ctx = probs @ V  (V fragments contiguous along key dim)
  for (int nt = 0; nt < 4; nt++) {
    v8f c = v8f{};
    int dh = nt * 16 + l15;             // B col
    const unsigned short* vrow = Vt + (((size_t)b * NH_ + h) * DH_ + dh) * S_;
    for (int kt = 0; kt < 32; kt++) {
      Frag ap, bv;
      #pragma unroll
      for (int i = 0; i < 8; i++) {
        ap.u[i] = *(const uint32_t*)&sp[l15 * SPS + kt * 32 + kpatA(i, hi)];
        bv.u[i] = *(const uint32_t*)(vrow + kt * 32 + kpatB(i, hi));
      }
      c = wmma_bf16(ap, bv, c);
    }
    #pragma unroll
    for (int j = 0; j < 8; j++)
      ctx[((size_t)(b * S_ + qt * 16 + j + 8 * hi)) * H_ + h * DH_ + nt * 16 + l15] = c[j];
  }
  #undef TOKBASE
}

// ---------------------------------------------------------------------------
// LayerNorm over H=768, one block per token (works in place).
// ---------------------------------------------------------------------------
__global__ __launch_bounds__(256) void ln_kernel(
    const float* __restrict__ x, const float* __restrict__ g,
    const float* __restrict__ bta, float* __restrict__ y)
{
  __shared__ float red[256];
  const int t = blockIdx.x, tid = threadIdx.x;
  const float* row = x + (size_t)t * H_;
  float v[3], s = 0.f;
  #pragma unroll
  for (int i = 0; i < 3; i++) { v[i] = row[tid + i * 256]; s += v[i]; }
  red[tid] = s; __syncthreads();
  for (int o = 128; o > 0; o >>= 1) { if (tid < o) red[tid] += red[tid + o]; __syncthreads(); }
  float mean = red[0] / (float)H_;
  __syncthreads();
  float q = 0.f;
  #pragma unroll
  for (int i = 0; i < 3; i++) { float d = v[i] - mean; q += d * d; }
  red[tid] = q; __syncthreads();
  for (int o = 128; o > 0; o >>= 1) { if (tid < o) red[tid] += red[tid + o]; __syncthreads(); }
  float inv = rsqrtf(red[0] / (float)H_ + 1e-12f);
  #pragma unroll
  for (int i = 0; i < 3; i++) {
    int c = tid + i * 256;
    y[(size_t)t * H_ + c] = g[c] * (v[i] - mean) * inv + bta[c];
  }
}

// ---------------------------------------------------------------------------
// Router: 1 wave per token; dot over H for E=8 experts, softmax, argmax.
// ---------------------------------------------------------------------------
__global__ __launch_bounds__(256) void router_kernel(
    const float* __restrict__ attn, const float* __restrict__ Wr,
    const float* __restrict__ br, float* __restrict__ scores_out,
    int* __restrict__ expert_id, int* __restrict__ counts)
{
  const int lane = threadIdx.x & 31;
  const int t = blockIdx.x * 8 + (threadIdx.x >> 5);
  float p[E_];
  #pragma unroll
  for (int e = 0; e < E_; e++) p[e] = 0.f;
  for (int hh = lane; hh < H_; hh += 32) {
    float xv = attn[(size_t)t * H_ + hh];
    #pragma unroll
    for (int e = 0; e < E_; e++) p[e] += xv * Wr[hh * E_ + e];
  }
  #pragma unroll
  for (int e = 0; e < E_; e++)
    for (int o = 16; o > 0; o >>= 1) p[e] += __shfl_xor(p[e], o, 32);
  if (lane == 0) {
    float m = -3.4e38f;
    #pragma unroll
    for (int e = 0; e < E_; e++) { p[e] += br[e]; m = fmaxf(m, p[e]); }
    float s = 0.f;
    #pragma unroll
    for (int e = 0; e < E_; e++) { p[e] = __expf(p[e] - m); s += p[e]; }
    float inv = 1.0f / s;
    int arg = 0; float best = -1.f;
    #pragma unroll
    for (int e = 0; e < E_; e++) {
      p[e] *= inv;
      scores_out[(size_t)t * E_ + e] = p[e];
      if (p[e] > best) { best = p[e]; arg = e; }   // first max kept
    }
    expert_id[t] = arg;
    atomicAdd(&counts[arg], 1);
  }
}

__global__ void zero_counts_kernel(int* counts) {
  if (threadIdx.x < E_) counts[threadIdx.x] = 0;
}

__global__ void scan_kernel(const int* __restrict__ counts,
                            int* __restrict__ offsets, int* __restrict__ cursor) {
  if (threadIdx.x == 0) {
    int acc = 0;
    for (int e = 0; e < E_; e++) { offsets[e] = acc; acc += counts[e]; cursor[e] = 0; }
  }
}

__global__ __launch_bounds__(256) void assign_kernel(
    const int* __restrict__ expert_id, const int* __restrict__ offsets,
    int* __restrict__ cursor, int* __restrict__ tok_of_pos)
{
  int t = blockIdx.x * 256 + threadIdx.x;
  if (t < T_) {
    int e = expert_id[t];
    int pos = offsets[e] + atomicAdd(&cursor[e], 1);
    tok_of_pos[pos] = t;
  }
}

// ---------------------------------------------------------------------------
extern "C" void kernel_launch(void* const* d_in, const int* in_sizes, int n_in,
                              void* d_out, int out_size, void* d_ws, size_t ws_size,
                              hipStream_t stream) {
  const float* X    = (const float*)d_in[0];
  const float* Wq   = (const float*)d_in[1];  const float* bq  = (const float*)d_in[2];
  const float* Wk   = (const float*)d_in[3];  const float* bk  = (const float*)d_in[4];
  const float* Wv   = (const float*)d_in[5];  const float* bv  = (const float*)d_in[6];
  const float* Wao  = (const float*)d_in[7];  const float* bao = (const float*)d_in[8];
  const float* ln1g = (const float*)d_in[9];  const float* ln1b = (const float*)d_in[10];
  const float* Wr   = (const float*)d_in[11]; const float* br  = (const float*)d_in[12];
  const float* Wi   = (const float*)d_in[13]; const float* bi  = (const float*)d_in[14];
  const float* Wo   = (const float*)d_in[15]; const float* bo  = (const float*)d_in[16];
  const float* ln2g = (const float*)d_in[17]; const float* ln2b = (const float*)d_in[18];

  float* out_layer  = (float*)d_out;                      // [T, H]
  float* out_scores = (float*)d_out + (size_t)T_ * H_;    // [T, E]

  // workspace carve-out
  char* w = (char*)d_ws;
  size_t off = 0;
  auto carve = [&](size_t bytes) -> char* {
    char* p = w + off; off = (off + bytes + 255) & ~(size_t)255; return p;
  };
  unsigned short* Qb   = (unsigned short*)carve((size_t)T_ * H_ * 2);
  unsigned short* Kb   = (unsigned short*)carve((size_t)T_ * H_ * 2);
  unsigned short* Vt   = (unsigned short*)carve((size_t)T_ * H_ * 2); // [B][NH][DH][S]
  float*          ctx  = (float*)carve((size_t)T_ * H_ * 4);
  float*          attn = (float*)carve((size_t)T_ * H_ * 4);   // pre-LN then LN (in place)
  unsigned short* intg = (unsigned short*)carve((size_t)T_ * F_ * 2);
  float*          outp = (float*)carve((size_t)T_ * H_ * 4);
  unsigned short* WqT  = (unsigned short*)carve((size_t)H_ * H_ * 2);
  unsigned short* WkT  = (unsigned short*)carve((size_t)H_ * H_ * 2);
  unsigned short* WvT  = (unsigned short*)carve((size_t)H_ * H_ * 2);
  unsigned short* WaoT = (unsigned short*)carve((size_t)H_ * H_ * 2);
  unsigned short* WiT  = (unsigned short*)carve((size_t)E_ * H_ * F_ * 2);
  unsigned short* WoT  = (unsigned short*)carve((size_t)F_ * H_ * 2);
  int* expert_id  = (int*)carve((size_t)T_ * 4);
  int* tok_of_pos = (int*)carve((size_t)T_ * 4);
  int* counts     = (int*)carve(E_ * 4);
  int* offsets    = (int*)carve(E_ * 4);
  int* cursor     = (int*)carve(E_ * 4);
  (void)in_sizes; (void)n_in; (void)out_size; (void)ws_size;

  dim3 blk(256);

  // 0) one-time weight transpose+convert to bf16 [N][K]
  transpose_bf16_kernel<<<dim3(H_/32, H_/32, 1), blk, 0, stream>>>(Wq,  WqT,  H_, H_);
  transpose_bf16_kernel<<<dim3(H_/32, H_/32, 1), blk, 0, stream>>>(Wk,  WkT,  H_, H_);
  transpose_bf16_kernel<<<dim3(H_/32, H_/32, 1), blk, 0, stream>>>(Wv,  WvT,  H_, H_);
  transpose_bf16_kernel<<<dim3(H_/32, H_/32, 1), blk, 0, stream>>>(Wao, WaoT, H_, H_);
  transpose_bf16_kernel<<<dim3(F_/32, H_/32, E_), blk, 0, stream>>>(Wi, WiT, H_, F_);
  transpose_bf16_kernel<<<dim3(H_/32, F_/32, 1), blk, 0, stream>>>(Wo, WoT, F_, H_);

  dim3 gQKV(H_ / BN, T_ / BM, 1);        // (6, 128)

  // 1) Q,K bf16 [T,H]; V bf16 in [B][NH][DH][S] (OUTM=2)
  wmma_gemm_kernel<float, 0, false, 1, false, false, false>
      <<<gQKV, blk, 0, stream>>>(X, WqT, bq, nullptr, Qb, nullptr, nullptr, nullptr,
                                 T_, H_, H_, 0, 0);
  wmma_gemm_kernel<float, 0, false, 1, false, false, false>
      <<<gQKV, blk, 0, stream>>>(X, WkT, bk, nullptr, Kb, nullptr, nullptr, nullptr,
                                 T_, H_, H_, 0, 0);
  wmma_gemm_kernel<float, 0, false, 2, false, false, false>
      <<<gQKV, blk, 0, stream>>>(X, WvT, bv, nullptr, Vt, nullptr, nullptr, nullptr,
                                 T_, H_, H_, 0, 0);

  // 2) attention -> ctx
  attn_kernel<<<dim3(S_ / 16, NH_, B_), dim3(32), 0, stream>>>(Qb, Kb, Vt, ctx);

  // 3) attn_pre = ctx @ Wao + bao + X ; LN1 in place
  wmma_gemm_kernel<float, 0, true, 0, false, false, false>
      <<<gQKV, blk, 0, stream>>>(ctx, WaoT, bao, X, attn, nullptr, nullptr, nullptr,
                                 T_, H_, H_, 0, 0);
  ln_kernel<<<dim3(T_), blk, 0, stream>>>(attn, ln1g, ln1b, attn);

  // 4) router -> scores (d_out), expert ids, counts; scan; assign permutation
  zero_counts_kernel<<<dim3(1), dim3(32), 0, stream>>>(counts);
  router_kernel<<<dim3(T_ / 8), blk, 0, stream>>>(attn, Wr, br, out_scores,
                                                  expert_id, counts);
  scan_kernel<<<dim3(1), dim3(32), 0, stream>>>(counts, offsets, cursor);
  assign_kernel<<<dim3(T_ / 256), blk, 0, stream>>>(expert_id, offsets, cursor,
                                                    tok_of_pos);

  // 5) FFN1 (per-expert segments, gather A rows): gelu(attn[idx] @ Wi[e] + bi[e])
  wmma_gemm_kernel<float, 1, false, 1, true, false, true>
      <<<dim3(F_ / BN, T_ / BM, E_), blk, 0, stream>>>(
          attn, WiT, bi, nullptr, intg, tok_of_pos, offsets, counts,
          T_, F_, H_, (long)H_ * F_, F_);

  // 6) FFN2 (dense over gathered rows, scatter C + residual by token)
  wmma_gemm_kernel<unsigned short, 0, true, 0, false, true, false>
      <<<dim3(H_ / BN, T_ / BM, 1), blk, 0, stream>>>(
          intg, WoT, bo, attn, outp, tok_of_pos, nullptr, nullptr,
          T_, H_, F_, 0, 0);

  // 7) LN2 -> layer_output
  ln_kernel<<<dim3(T_), blk, 0, stream>>>(outp, ln2g, ln2b, out_layer);
}
